// TensorConvLayer_7627861918027
// MI455X (gfx1250) — matrix operand order repaired and checked by reading, not compiled, not run on hardware
//
#include <hip/hip_runtime.h>
#include <math.h>

// ---------------------------------------------------------------------------
// TensorConvLayer for MI455X (gfx1250, wave32).
// Fused edge kernel: per wave = 16 edges.
//   MLP1 (16x64 @ 64x64) and MLP2 (16x64 @ 64x576) via v_wmma_f32_16x16x32_bf16.
//   Weights are pre-swizzled ONCE (tcl_prep_w) into a global bf16 image laid
//   out exactly as the B-fragment register layout; each workgroup then pulls
//   the 80KB image into LDS with a flat async copy
//   (global_load_async_to_lds_b128). h is stored pre-swizzled into the
//   A-fragment layout. Per-edge tensor-product factors are stored transposed
//   [u][e] so consumption is a packed v8f FMA against the WMMA accumulator,
//   straight from the C-layout registers (w never hits memory).
// Scatter: global f32 atomics into segment sums; node kernels do mean/residual
// and the scalar/vector batch-norms.
// ---------------------------------------------------------------------------

typedef __attribute__((ext_vector_type(16))) __bf16 v16bf;
typedef __attribute__((ext_vector_type(8)))  float  v8f;

static __device__ inline v8f wmma_bf16(v16bf a, v16bf b, v8f c) {
  // 8 args: (neg_a, A, neg_b, B, c_mod, C, reuse_a, reuse_b)
  return __builtin_amdgcn_wmma_f32_16x16x32_bf16(false, a, false, b, (short)0, c,
                                                 false, false);
}

static __device__ inline v8f splat8(float x) {
  v8f v;
#pragma unroll
  for (int r = 0; r < 8; ++r) v[r] = x;
  return v;
}

// Swizzled B fragment: 16 contiguous bf16 per (tileIdx, lane) -> one 32B load.
static __device__ inline v16bf load_b_frag_sw(const __bf16* __restrict__ base,
                                              int tileIdx, int lane) {
  return *(const v16bf*)(base + ((tileIdx * 32) + lane) * 16);
}

// ---- LDS layout (bytes), dynamic shared memory -----------------------------
// Swizzled weights: element j of fragment (t, s, lane) lives at
//   [((t*2+s)*32 + lane)*16 + j]
#define OFF_W1   0        // 64x64  bf16 swizzled =  8192
#define OFF_W2   8192     // 64x576 bf16 swizzled = 73728
#define W_BYTES  81920    // flat weight image (matches global wpre image)
#define OFF_B1   81920    // 64  f32
#define OFF_B2   82176    // 576 f32
#define OFF_PW   84480    // per-wave blocks
#define PW_H     0        // h, A-fragment-swizzled: (s*32+lane)*16+j bf16 = 2048
#define PW_X0S   2048     // [u][e] 16x16 f32 = 1024   (x0 * sh0)
#define PW_X0    3072     // [u][e] 16x16 f32 = 1024
#define PW_A1    4096     // [u][e]  8x16 f32 =  512   (inv3 * x1.sh1)
#define PW_X1    4608     // [i][u][e] 3x8x16 f32 = 1536
#define PW_SH    6144     // 16x4  f32  =  256
#define PW_SRC   6400     // 16 int     =   64
#define PW_STRIDE 6464
#define SMEM_TOTAL (OFF_PW + 8 * PW_STRIDE)   // 136192 B < 320KB WGP LDS

#if __has_builtin(__builtin_amdgcn_global_load_async_to_lds_b128)
#define TCL_ASYNC_LDS 1
// Exact parameter types per compiler diagnostic: int4 (gcc vector) pointers in
// global (addrspace 1) and LDS (addrspace 3) address spaces.
typedef __attribute__((__vector_size__(4 * sizeof(int)))) int tcl_vi4;
typedef __attribute__((address_space(1))) tcl_vi4* tcl_gv4p;
typedef __attribute__((address_space(3))) tcl_vi4* tcl_lv4p;
#endif

static __device__ inline void tcl_wait_async0() {
#if __has_builtin(__builtin_amdgcn_s_wait_asynccnt)
  __builtin_amdgcn_s_wait_asynccnt(0);
#else
  asm volatile("s_wait_asynccnt 0" ::: "memory");
#endif
}

// ---- one-time weight pre-swizzle (f32 -> bf16 fragment image) --------------
__global__ void tcl_prep_w(const float* __restrict__ w1g,
                           const float* __restrict__ w2g,
                           __bf16* __restrict__ wpre) {
  int i = blockIdx.x * blockDim.x + threadIdx.x;
  if (i < 4096) {            // W1 (64x64), source-linear
    int k = i >> 6, c = i & 63;
    int t = c >> 4, n = c & 15;
    int s = k >> 5, hf = (k >> 3) & 1, j = (k & 7) + (((k >> 4) & 1) << 3);
    wpre[(((t * 2 + s) * 32) + hf * 16 + n) * 16 + j] = (__bf16)w1g[i];
  } else if (i < 4096 + 36864) {  // W2 (64x576), source-linear
    int ii = i - 4096;
    int k = ii / 576, c = ii - k * 576;
    int t = c >> 4, n = c & 15;
    int s = k >> 5, hf = (k >> 3) & 1, j = (k & 7) + (((k >> 4) & 1) << 3);
    wpre[4096 + (((t * 2 + s) * 32) + hf * 16 + n) * 16 + j] = (__bf16)w2g[ii];
  }
}

__global__ __launch_bounds__(256)
void tcl_edge_kernel(const float* __restrict__ nf,
                     const float* __restrict__ ef,
                     const float* __restrict__ esh,
                     const int*   __restrict__ eidx,
                     const __bf16* __restrict__ wpre,
                     const float* __restrict__ b1g,
                     const float* __restrict__ b2g,
                     float* __restrict__ seg,
                     float* __restrict__ cnt,
                     int E) {
  extern __shared__ char smem[];
  __bf16* sW1 = (__bf16*)(smem + OFF_W1);
  __bf16* sW2 = (__bf16*)(smem + OFF_W2);
  float*  sB1 = (float*)(smem + OFF_B1);
  float*  sB2 = (float*)(smem + OFF_B2);

  const int tid  = threadIdx.x;
  const int lane = tid & 31;
  const int half = lane >> 4;
  const int nn   = lane & 15;
  const int wv   = tid >> 5;
  const int tileBase = blockIdx.x * 128 + wv * 16;

  char* pw = smem + OFF_PW + wv * PW_STRIDE;
  __bf16* sHs  = (__bf16*)(pw + PW_H);
  float*  sX0s = (float*)(pw + PW_X0S);
  float*  sX0  = (float*)(pw + PW_X0);
  float*  sA1  = (float*)(pw + PW_A1);
  float*  sX1  = (float*)(pw + PW_X1);
  float*  sSh  = (float*)(pw + PW_SH);
  int*    sSrc = (int*)(pw + PW_SRC);

  // Prefetch next block's edge-feature tile.
  if (tileBase + 128 < E)
    __builtin_prefetch(ef + (size_t)(tileBase + 128) * 64, 0, 0);

  // Flat 80KB copy of the pre-swizzled weight image into LDS.
  {
    const char* wsrc = (const char*)wpre;
#ifdef TCL_ASYNC_LDS
    for (int i = tid; i < W_BYTES / 16; i += 256)
      __builtin_amdgcn_global_load_async_to_lds_b128(
          (tcl_gv4p)(wsrc + (size_t)i * 16), (tcl_lv4p)(smem + i * 16), 0, 0);
#else
    for (int i = tid; i < W_BYTES / 16; i += 256)
      ((int4*)smem)[i] = ((const int4*)wsrc)[i];
#endif
  }
  if (tid < 64) sB1[tid] = b1g[tid];
  for (int i = tid; i < 576; i += 256) sB2[i] = b2g[i];

  // Per-edge tensor-product operands (transposed [u][e]); lanes 0..15 = edges.
  const float inv3 = 0.57735026918962576f;  // 1/sqrt(3)
  if (lane < 16) {
    int e = tileBase + lane;
    if (e < E) {
      int dst = eidx[e];
      int src = eidx[E + e];
      sSrc[lane] = src;
      float s0 = esh[(size_t)e * 4 + 0];
      float s1 = esh[(size_t)e * 4 + 1];
      float s2 = esh[(size_t)e * 4 + 2];
      float s3 = esh[(size_t)e * 4 + 3];
      sSh[lane * 4 + 0] = s0; sSh[lane * 4 + 1] = s1;
      sSh[lane * 4 + 2] = s2; sSh[lane * 4 + 3] = s3;
      const float* xp = nf + (size_t)dst * 40;
#pragma unroll
      for (int u = 0; u < 16; ++u) {
        float x = xp[u];
        sX0[u * 16 + lane]  = x;
        sX0s[u * 16 + lane] = x * s0;
      }
#pragma unroll
      for (int u = 0; u < 8; ++u) {
        float xa = xp[16 + u * 3 + 0];
        float xb = xp[16 + u * 3 + 1];
        float xc = xp[16 + u * 3 + 2];
        sX1[(0 * 8 + u) * 16 + lane] = xa;
        sX1[(1 * 8 + u) * 16 + lane] = xb;
        sX1[(2 * 8 + u) * 16 + lane] = xc;
        sA1[u * 16 + lane] = inv3 * (xa * s1 + xb * s2 + xc * s3);
      }
      atomicAdd(&cnt[src], 1.0f);
    } else {
      sSrc[lane] = -1;
#pragma unroll
      for (int u = 0; u < 16; ++u) { sX0[u * 16 + lane] = 0.f; sX0s[u * 16 + lane] = 0.f; }
#pragma unroll
      for (int u = 0; u < 8; ++u) {
        sA1[u * 16 + lane] = 0.f;
        sX1[(0 * 8 + u) * 16 + lane] = 0.f;
        sX1[(1 * 8 + u) * 16 + lane] = 0.f;
        sX1[(2 * 8 + u) * 16 + lane] = 0.f;
      }
      sSh[lane * 4 + 0] = 0.f; sSh[lane * 4 + 1] = 0.f;
      sSh[lane * 4 + 2] = 0.f; sSh[lane * 4 + 3] = 0.f;
    }
  }
#ifdef TCL_ASYNC_LDS
  tcl_wait_async0();  // each wave drains its own async copies before barrier
#endif
  __syncthreads();

  // A fragments (16x32 bf16, two K-steps) straight from global edge features.
  v16bf a0, a1v;
  {
    int row = tileBase + nn;
    if (row >= E) row = E - 1;
    const float* ep = ef + (size_t)row * 64;
#pragma unroll
    for (int j = 0; j < 16; ++j) {
      int kk = 8 * half + (j < 8 ? j : j + 8);
      a0[j]  = (__bf16)ep[kk];
      a1v[j] = (__bf16)ep[32 + kk];
    }
  }

  // MLP1: h = relu(ef @ W1 + b1); store h A-fragment-swizzled into LDS.
#pragma unroll
  for (int t = 0; t < 4; ++t) {
    v8f acc = splat8(sB1[16 * t + nn]);
    v16bf bf0 = load_b_frag_sw(sW1, t * 2 + 0, lane);
    v16bf bf1 = load_b_frag_sw(sW1, t * 2 + 1, lane);
    acc = wmma_bf16(a0, bf0, acc);
    acc = wmma_bf16(a1v, bf1, acc);
    // Element (m = r + 8*half, col c = 16t+nn) -> A-frag (s, lane', j):
    //   s = c>>5, kk = c&31, lane' = ((kk>>3)&1)*16 + m, j = (kk&7) + (kk>=16?8:0)
    int c  = 16 * t + nn;
    int s  = c >> 5, kk = c & 31;
    int hf2 = (kk >> 3) & 1;
    int j2  = (kk & 7) + ((kk >> 4) << 3);
    __bf16* dstp = sHs + (s * 32 + hf2 * 16 + 8 * half) * 16 + j2;
#pragma unroll
    for (int r = 0; r < 8; ++r)
      dstp[r * 16] = (__bf16)fmaxf(acc[r], 0.0f);
  }
  __syncthreads();

  // A fragments for MLP2: one 32B vector LDS load per K-step.
  v16bf c0  = *(const v16bf*)(sHs + (0 * 32 + lane) * 16);
  v16bf c1v = *(const v16bf*)(sHs + (1 * 32 + lane) * 16);

  // Tensor-product accumulators, per (edge r, channel nn).
  v8f out0acc = splat8(0.f);  // out0[v=nn], sh0 & inv3 pre-folded
  v8f t01acc  = splat8(0.f);  // partial sum_u x0[u] w01[u, v=nn&7]
  v8f o1bx = splat8(0.f), o1by = splat8(0.f), o1bz = splat8(0.f);

  const int eb = 8 * half;  // first edge owned by this lane-half

  // w00[u=t, v=nn], t = 0..15
#pragma unroll 4
  for (int t = 0; t < 16; ++t) {
    v8f acc = splat8(sB2[16 * t + nn]);
    acc = wmma_bf16(c0,  load_b_frag_sw(sW2, t * 2 + 0, lane), acc);
    acc = wmma_bf16(c1v, load_b_frag_sw(sW2, t * 2 + 1, lane), acc);
    v8f xv = *(const v8f*)(sX0s + t * 16 + eb);
    out0acc += xv * acc;
  }
  // w11[u, v=nn], t = 16..23
#pragma unroll 4
  for (int u = 0; u < 8; ++u) {
    int t = 16 + u;
    v8f acc = splat8(sB2[16 * t + nn]);
    acc = wmma_bf16(c0,  load_b_frag_sw(sW2, t * 2 + 0, lane), acc);
    acc = wmma_bf16(c1v, load_b_frag_sw(sW2, t * 2 + 1, lane), acc);
    v8f xv = *(const v8f*)(sA1 + u * 16 + eb);
    out0acc += xv * acc;
  }
  // w01[u, v=nn&7], t = 24..31, u = 2(t-24)+(nn>>3)
#pragma unroll 4
  for (int tt = 0; tt < 8; ++tt) {
    int t = 24 + tt;
    int u = 2 * tt + (nn >> 3);
    v8f acc = splat8(sB2[16 * t + nn]);
    acc = wmma_bf16(c0,  load_b_frag_sw(sW2, t * 2 + 0, lane), acc);
    acc = wmma_bf16(c1v, load_b_frag_sw(sW2, t * 2 + 1, lane), acc);
    v8f xv = *(const v8f*)(sX0 + u * 16 + eb);
    t01acc += xv * acc;
  }
  // w10[u, v=nn&7], t = 32..35, u = 2(t-32)+(nn>>3)
#pragma unroll 2
  for (int tt = 0; tt < 4; ++tt) {
    int t = 32 + tt;
    int u = 2 * tt + (nn >> 3);
    v8f acc = splat8(sB2[16 * t + nn]);
    acc = wmma_bf16(c0,  load_b_frag_sw(sW2, t * 2 + 0, lane), acc);
    acc = wmma_bf16(c1v, load_b_frag_sw(sW2, t * 2 + 1, lane), acc);
    o1bx += *(const v8f*)(sX1 + (0 * 8 + u) * 16 + eb) * acc;
    o1by += *(const v8f*)(sX1 + (1 * 8 + u) * 16 + eb) * acc;
    o1bz += *(const v8f*)(sX1 + (2 * 8 + u) * 16 + eb) * acc;
  }

  // Scatter into segment sums (linear, so split lane-half partials just add).
  const float alpha = 0.20412414523193151f;  // 1/sqrt(M0+M1)
  const int vv = nn & 7;
#pragma unroll
  for (int r = 0; r < 8; ++r) {
    int el = eb + r;
    int node = sSrc[el];
    if (node < 0) continue;
    float* segn = seg + (size_t)node * 40;
    atomicAdd(segn + nn, alpha * out0acc[r]);   // unique (e, v) per lane
    float s0 = sSh[el * 4 + 0];
    float o1[3] = { o1bx[r], o1by[r], o1bz[r] };
#pragma unroll
    for (int i = 0; i < 3; ++i) {
      float val = alpha * (sSh[el * 4 + 1 + i] * t01acc[r] + s0 * o1[i]);
      atomicAdd(segn + 16 + vv * 3 + i, val);
    }
  }
}

// ---- node-level kernels ----------------------------------------------------

__global__ void tcl_zero(float* __restrict__ ws, int n1, int gap, int n2) {
  int i = blockIdx.x * blockDim.x + threadIdx.x;
  if (i < n1) ws[i] = 0.0f;
  else if (i < n1 + n2) ws[gap + (i - n1)] = 0.0f;
}

__global__ void tcl_node_pre(const float* __restrict__ nf,
                             const float* __restrict__ seg,
                             const float* __restrict__ cnt,
                             float* __restrict__ pre,
                             float* __restrict__ part, int N) {
  __shared__ float red[40];
  int tid = threadIdx.x;
  if (tid < 40) red[tid] = 0.0f;
  __syncthreads();
  int node = blockIdx.x * blockDim.x + tid;
  if (node < N) {
    float inv = 1.0f / fmaxf(cnt[node], 1.0f);
    const float* sp = seg + (size_t)node * 40;
    const float* np = nf + (size_t)node * 40;
    float* pp = pre + (size_t)node * 40;
    float p[40];
#pragma unroll
    for (int c = 0; c < 40; ++c) { p[c] = sp[c] * inv + np[c]; pp[c] = p[c]; }
#pragma unroll
    for (int c = 0; c < 16; ++c) {
      atomicAdd(&red[c], p[c]);
      atomicAdd(&red[16 + c], p[c] * p[c]);
    }
#pragma unroll
    for (int u = 0; u < 8; ++u) {
      float a = p[16 + u * 3], b = p[17 + u * 3], c2 = p[18 + u * 3];
      atomicAdd(&red[32 + u], a * a + b * b + c2 * c2);
    }
  }
  __syncthreads();
  if (tid < 40) atomicAdd(&part[tid], red[tid]);
}

__global__ void tcl_finalize(const float* __restrict__ part,
                             const float* __restrict__ bnws,
                             const float* __restrict__ bnbs,
                             const float* __restrict__ bnwv,
                             float* __restrict__ fact, int N) {
  int tid = threadIdx.x;
  float fN = (float)N;
  if (tid < 16) {
    float mu  = part[tid] / fN;
    float var = part[16 + tid] / fN - mu * mu;
    float sc  = rsqrtf(var + 1e-5f) * bnws[tid];
    fact[tid] = sc;
    fact[16 + tid] = bnbs[tid] - mu * sc;
  } else if (tid >= 32 && tid < 40) {
    float vn = part[tid] / (3.0f * fN);
    fact[tid] = rsqrtf(vn + 1e-5f) * bnwv[tid - 32];
  }
}

__global__ void tcl_apply(const float* __restrict__ pre,
                          const float* __restrict__ fact,
                          float* __restrict__ out, int total) {
  int i = blockIdx.x * blockDim.x + threadIdx.x;
  if (i >= total) return;
  int ch = i % 40;
  float p = pre[i];
  float o;
  if (ch < 16) o = p * fact[ch] + fact[16 + ch];
  else         o = p * fact[32 + (ch - 16) / 3];
  out[i] = o;
}

// ---------------------------------------------------------------------------

extern "C" void kernel_launch(void* const* d_in, const int* in_sizes, int n_in,
                              void* d_out, int out_size, void* d_ws, size_t ws_size,
                              hipStream_t stream) {
  const float* nf   = (const float*)d_in[0];
  const float* ef   = (const float*)d_in[1];
  const float* esh  = (const float*)d_in[2];
  const int*   eix  = (const int*)d_in[3];
  const float* w1   = (const float*)d_in[4];
  const float* b1   = (const float*)d_in[5];
  const float* w2   = (const float*)d_in[6];
  const float* b2   = (const float*)d_in[7];
  const float* bnws = (const float*)d_in[8];
  const float* bnbs = (const float*)d_in[9];
  const float* bnwv = (const float*)d_in[10];
  float* out = (float*)d_out;
  float* ws  = (float*)d_ws;

  const int N = in_sizes[0] / 40;
  const int E = in_sizes[1] / 64;

  float*  seg  = ws;                       // N*40
  float*  cnt  = ws + (size_t)N * 40;      // N
  float*  pre  = ws + (size_t)N * 41;      // N*40
  float*  part = ws + (size_t)N * 81;      // 40 (sum_s, sumsq_s, sumv)
  float*  fact = part + 40;                // 40 (scale, shift, vscale)
  __bf16* wpre = (__bf16*)(fact + 40);     // 81920 B pre-swizzled weight image

  {
    int tot = N * 41 + 40;
    tcl_zero<<<(tot + 255) / 256, 256, 0, stream>>>(ws, N * 41, N * 81, 40);
  }
  tcl_prep_w<<<(4096 + 36864 + 255) / 256, 256, 0, stream>>>(w1, w2, wpre);
  {
    int nblk = (E + 127) / 128;  // 128 edges per workgroup (8 waves x 16)
    tcl_edge_kernel<<<nblk, 256, SMEM_TOTAL, stream>>>(
        nf, ef, esh, eix, wpre, b1, b2, seg, cnt, E);
  }
  tcl_node_pre<<<(N + 255) / 256, 256, 0, stream>>>(nf, seg, cnt, pre, part, N);
  tcl_finalize<<<1, 64, 0, stream>>>(part, bnws, bnbs, bnwv, fact, N);
  {
    int tot = N * 40;
    tcl_apply<<<(tot + 255) / 256, 256, 0, stream>>>(pre, fact, out, tot);
  }
  // Second tuple output: edge_features passthrough.
  (void)hipMemcpyAsync(out + (size_t)N * 40, ef, (size_t)E * 64 * sizeof(float),
                       hipMemcpyDeviceToDevice, stream);
}
